// CausalAttention_67637144977575
// MI455X (gfx1250) — matrix operand order repaired and checked by reading, not compiled, not run on hardware
//
#include <hip/hip_runtime.h>
#include <hip/hip_bf16.h>

typedef __attribute__((ext_vector_type(16))) _Float16 v16h;
typedef __attribute__((ext_vector_type(8)))  _Float16 v8h;
typedef __attribute__((ext_vector_type(8)))  float    v8f;
typedef __attribute__((ext_vector_type(4)))  float    f32x4;

#define WMMA_F16(a, b, c) \
    __builtin_amdgcn_wmma_f32_16x16x32_f16(false, (a), false, (b), (short)0, (c), false, false)

union V16H {
    v16h v;
    v8h  h8[2];
    _Float16 e[16];
};

// ---- Fragment loaders (layouts per CDNA5 ISA 7.12.2) ----

// A-fragment (16x32 f16, M x K) from a row-major f16 matrix.
// base points at tile element [0][0]; stride = row stride in elements.
// lane L: row = L&15;  halves 0..7 -> k = 8*(L>>4)+0..7, halves 8..15 -> +16.
__device__ inline v16h load_a_f16(const _Float16* base, int stride, int lane) {
    int m = lane & 15, g = lane >> 4;
    const _Float16* p = base + (size_t)m * stride + 8 * g;
    V16H u;
    u.h8[0] = *(const v8h*)(p);
    u.h8[1] = *(const v8h*)(p + 16);
    return u.v;
}

// A-fragment from a row-major fp32 matrix, converting to f16.
__device__ inline v16h load_a_f32cvt(const float* base, int stride, int lane) {
    int m = lane & 15, g = lane >> 4;
    const f32x4* p4 = (const f32x4*)(base + (size_t)m * stride + 8 * g);
    f32x4 c0 = p4[0], c1 = p4[1], c2 = p4[4], c3 = p4[5];
    V16H u;
#pragma unroll
    for (int i = 0; i < 4; ++i) {
        u.e[i]      = (_Float16)c0[i];
        u.e[4 + i]  = (_Float16)c1[i];
        u.e[8 + i]  = (_Float16)c2[i];
        u.e[12 + i] = (_Float16)c3[i];
    }
    return u.v;
}

// B-fragment (32x16 f16, K x N) from B^T stored row-major as [N][K].
// lane L: col n = L&15;  halves 0..7 -> k = 8*(L>>4)+0..7, halves 8..15 -> +16.
__device__ inline v16h load_bt_f16(const _Float16* base, int stride, int lane) {
    int n = lane & 15, g = lane >> 4;
    const _Float16* p = base + (size_t)n * stride + 8 * g;
    V16H u;
    u.h8[0] = *(const v8h*)(p);
    u.h8[1] = *(const v8h*)(p + 16);
    return u.v;
}

// ---------------- Kernel 0: weights -> f16, transposed [3][64][1024] ----------------
__global__ __launch_bounds__(256) void wtrans_kernel(const float* __restrict__ Wq,
                                                     const float* __restrict__ Wk,
                                                     const float* __restrict__ Wv,
                                                     _Float16* __restrict__ WT) {
    int t = blockIdx.x * 256 + threadIdx.x;      // 0 .. 3*64*1024-1
    int w = t >> 16;                             // which matrix
    int rem = t & 65535;
    int n = rem >> 10;                           // output feature 0..63
    int k = rem & 1023;                          // input feature 0..1023
    const float* W = (w == 0) ? Wq : (w == 1) ? Wk : Wv;
    WT[t] = (_Float16)W[(size_t)k * 64 + n];
}

// ---------------- Kernel 1: fused QKV projection ----------------
// One wave per 16-row tile of the flattened [B*S, 1024] input.
// Q,K stored f16 [B*S][64]; V stored transposed f16 [B][64][S].
__global__ __launch_bounds__(128) void qkv_kernel(const float* __restrict__ x,
                                                  const _Float16* __restrict__ WT,
                                                  const float* __restrict__ bq,
                                                  const float* __restrict__ bk,
                                                  const float* __restrict__ bv,
                                                  _Float16* __restrict__ Qb,
                                                  _Float16* __restrict__ Kb,
                                                  _Float16* __restrict__ VTb) {
    int lane = threadIdx.x & 31;
    int wv   = threadIdx.x >> 5;
    int rowtile = blockIdx.x * 4 + wv;           // 0..1023
    int rowbase = rowtile * 16;

    const _Float16* WTq = WT;
    const _Float16* WTk = WT + 65536;
    const _Float16* WTv = WT + 131072;

    v8f aq[4] = {}, ak[4] = {}, av[4] = {};

    for (int kt = 0; kt < 32; ++kt) {            // 32 x (k=32) = K 1024
        v16h a = load_a_f32cvt(x + (size_t)rowbase * 1024 + kt * 32, 1024, lane);
#pragma unroll
        for (int nt = 0; nt < 4; ++nt) {
            v16h b;
            b = load_bt_f16(WTq + (size_t)nt * 16 * 1024 + kt * 32, 1024, lane);
            aq[nt] = WMMA_F16(a, b, aq[nt]);
            b = load_bt_f16(WTk + (size_t)nt * 16 * 1024 + kt * 32, 1024, lane);
            ak[nt] = WMMA_F16(a, b, ak[nt]);
            b = load_bt_f16(WTv + (size_t)nt * 16 * 1024 + kt * 32, 1024, lane);
            av[nt] = WMMA_F16(a, b, av[nt]);
        }
    }

    int n0 = lane & 15, g = lane >> 4;
#pragma unroll
    for (int nt = 0; nt < 4; ++nt) {
        int n = nt * 16 + n0;
        float bqv = bq[n], bkv = bk[n], bvv = bv[n];
#pragma unroll
        for (int r = 0; r < 8; ++r) {
            int row = rowbase + r + 8 * g;       // C/D layout: row = r + 8*(lane>>4)
            Qb[(size_t)row * 64 + n] = (_Float16)(aq[nt][r] + bqv);
            Kb[(size_t)row * 64 + n] = (_Float16)(ak[nt][r] + bkv);
            int b_ = row >> 12, s_ = row & 4095;
            VTb[((size_t)(b_ * 64 + n)) * 4096 + s_] = (_Float16)(av[nt][r] + bvv);
        }
    }
}

// ---------------- Kernel 2: flash attention, one wave per 16-query tile ----------------
__global__ __launch_bounds__(32) void attn_kernel(const _Float16* __restrict__ Qb,
                                                  const _Float16* __restrict__ Kb,
                                                  const _Float16* __restrict__ VTb,
                                                  float* __restrict__ out) {
    __shared__ __align__(16) _Float16 Pbuf[16 * 32];   // prob tile staging

    const float SCALE = 0.015625f;                     // 1/sqrt(S) = 1/64
    const float LOG2E = 1.4426950408889634f;

    int lane  = threadIdx.x;
    int qtile = blockIdx.x;                            // 0..1023 over flattened [B*S/16]
    int qbase = qtile * 16;                            // flattened query row base
    int b     = qbase >> 12;
    int qloc  = qbase & 4095;                          // query base within the batch
    int n0 = lane & 15, g = lane >> 4;

    const _Float16* Kbatch = Kb  + (size_t)b * 4096 * 64;
    const _Float16* Vbatch = VTb + (size_t)b * 64 * 4096;

    // Q fragments for the two k32 halves of d_out=64 (kept in registers).
    v16h qf0 = load_a_f16(Qb + (size_t)qbase * 64,      64, lane);
    v16h qf1 = load_a_f16(Qb + (size_t)qbase * 64 + 32, 64, lane);

    float m[8], l[8], alpha[8];
    v8f acc[4] = {};
#pragma unroll
    for (int r = 0; r < 8; ++r) { m[r] = -3.0e38f; l[r] = 0.0f; }

    int nblk = (qloc + 16 + 31) >> 5;                  // # of 32-key blocks needed

    for (int j = 0; j < nblk; ++j) {
        int kb = j * 32;
        bool lastBlk = (j == nblk - 1);

        // Scores: two 16-key half tiles, K=64 reduction as 2 x k32.
        v8f s0 = {}, s1 = {};
        s0 = WMMA_F16(qf0, load_bt_f16(Kbatch + (size_t)(kb)      * 64,      64, lane), s0);
        s0 = WMMA_F16(qf1, load_bt_f16(Kbatch + (size_t)(kb)      * 64 + 32, 64, lane), s0);
        s1 = WMMA_F16(qf0, load_bt_f16(Kbatch + (size_t)(kb + 16) * 64,      64, lane), s1);
        s1 = WMMA_F16(qf1, load_bt_f16(Kbatch + (size_t)(kb + 16) * 64 + 32, 64, lane), s1);

        // Online softmax: per-row (query) stats via 16-lane shuffles.
#pragma unroll
        for (int r = 0; r < 8; ++r) {
            float v0 = s0[r] * SCALE;
            float v1 = s1[r] * SCALE;
            if (lastBlk) {
                int q = qloc + r + 8 * g;              // local query index of this element
                if (kb + n0      > q) v0 = -1.0e30f;   // causal mask
                if (kb + 16 + n0 > q) v1 = -1.0e30f;
            }
            float mx = fmaxf(v0, v1);
            mx = fmaxf(mx, __shfl_xor(mx, 1, 16));
            mx = fmaxf(mx, __shfl_xor(mx, 2, 16));
            mx = fmaxf(mx, __shfl_xor(mx, 4, 16));
            mx = fmaxf(mx, __shfl_xor(mx, 8, 16));
            float mn = fmaxf(m[r], mx);
            float al = exp2f((m[r] - mn) * LOG2E);
            float e0 = exp2f((v0 - mn) * LOG2E);
            float e1 = exp2f((v1 - mn) * LOG2E);
            float rs = e0 + e1;
            rs += __shfl_xor(rs, 1, 16);
            rs += __shfl_xor(rs, 2, 16);
            rs += __shfl_xor(rs, 4, 16);
            rs += __shfl_xor(rs, 8, 16);
            l[r] = l[r] * al + rs;
            m[r] = mn;
            alpha[r] = al;
            // stage probs row-major [16 rows][32 keys] for A-fragment reload
            Pbuf[(r + 8 * g) * 32 + n0]      = (_Float16)e0;
            Pbuf[(r + 8 * g) * 32 + 16 + n0] = (_Float16)e1;
        }
        __syncthreads();                               // single-wave WG: cheap; orders LDS
        v16h pf = load_a_f16(&Pbuf[0], 32, lane);

        // acc = alpha*acc + P @ V   (4 feature tiles, one k32 step over 32 keys)
#pragma unroll
        for (int nt = 0; nt < 4; ++nt) {
#pragma unroll
            for (int r = 0; r < 8; ++r) acc[nt][r] *= alpha[r];
            v16h vb = load_bt_f16(Vbatch + (size_t)nt * 16 * 4096 + kb, 4096, lane);
            acc[nt] = WMMA_F16(pf, vb, acc[nt]);
        }
        __syncthreads();
    }

    // out = acc / l   (fp32 output [B*S][64])
#pragma unroll
    for (int nt = 0; nt < 4; ++nt) {
        int n = nt * 16 + n0;
#pragma unroll
        for (int r = 0; r < 8; ++r) {
            int row = qbase + r + 8 * g;
            out[(size_t)row * 64 + n] = acc[nt][r] / l[r];
        }
    }
}

extern "C" void kernel_launch(void* const* d_in, const int* in_sizes, int n_in,
                              void* d_out, int out_size, void* d_ws, size_t ws_size,
                              hipStream_t stream) {
    (void)in_sizes; (void)n_in; (void)out_size; (void)ws_size;
    const float* x  = (const float*)d_in[0];
    const float* Wq = (const float*)d_in[1];
    const float* bq = (const float*)d_in[2];
    const float* Wk = (const float*)d_in[3];
    const float* bk = (const float*)d_in[4];
    const float* Wv = (const float*)d_in[5];
    const float* bv = (const float*)d_in[6];
    float* out = (float*)d_out;

    // Workspace layout (f16):
    //   WT : 3*64*1024        = 393216 B @ 0
    //   Q  : 16384*64*2       = 2 MiB   @ 393216
    //   K  : 2 MiB                      @ 2490368
    //   VT : 2 MiB                      @ 4587520   (total ~6.4 MiB)
    char* ws = (char*)d_ws;
    _Float16* WT  = (_Float16*)(ws);
    _Float16* Qb  = (_Float16*)(ws + 393216);
    _Float16* Kb  = (_Float16*)(ws + 2490368);
    _Float16* VTb = (_Float16*)(ws + 4587520);

    wtrans_kernel<<<768, 256, 0, stream>>>(Wq, Wk, Wv, WT);
    qkv_kernel<<<256, 128, 0, stream>>>(x, WT, bq, bk, bv, Qb, Kb, VTb);
    attn_kernel<<<1024, 32, 0, stream>>>(Qb, Kb, VTb, out);
}
